// Model_39676907887756
// MI455X (gfx1250) — matrix operand order repaired and checked by reading, not compiled, not run on hardware
//
#include <hip/hip_runtime.h>
#include <hip/hip_bf16.h>
#include <math.h>

typedef _Float16 v16h __attribute__((ext_vector_type(16)));
typedef _Float16 v8h  __attribute__((ext_vector_type(8)));
typedef float    v8f  __attribute__((ext_vector_type(8)));

#define B_    16
#define SQ_   2048
#define SKV_  2048
#define D_    128
#define BM    64     // Q rows per block (16 per wave x 4 waves)
#define BN    32     // KV columns per iteration
#define NWAVES 4
#define NIT   (SKV_ / BN)

// v_exp_f32 (base-2 exponent) — bypass libm, use the backend builtin directly.
__device__ __forceinline__ float exp2_fast(float x) {
    return __builtin_amdgcn_exp2f(x);
}

// ---- DPP-based cross-lane helpers (pure VALU; keeps DS pipe free) ----
#define DPP8SEL(a,b,c,d,e,f,g,h) \
    ((a)|((b)<<3)|((c)<<6)|((d)<<9)|((e)<<12)|((f)<<15)|((g)<<18)|((h)<<21))
#define DPP8_XOR1 DPP8SEL(1,0,3,2,5,4,7,6)
#define DPP8_XOR2 DPP8SEL(2,3,0,1,6,7,4,5)
#define DPP8_XOR4 DPP8SEL(4,5,6,7,0,1,2,3)

template <int SEL>
__device__ __forceinline__ float dpp8f(float x) {
    return __int_as_float(__builtin_amdgcn_mov_dpp8(__float_as_int(x), SEL));
}
// DPP16 row_mirror (ctrl 0x140): lane -> 15-lane within each 16-lane row.
__device__ __forceinline__ float row_mirror(float x) {
    return __int_as_float(
        __builtin_amdgcn_mov_dpp(__float_as_int(x), 0x140, 0xf, 0xf, true));
}
// Reduce over the 16 lanes of each half-wave row (xor1/2/4 inside the 8-group,
// then row_mirror to combine the two 8-groups — valid once groups are uniform).
__device__ __forceinline__ float rowmax16(float t) {
    t = fmaxf(t, dpp8f<DPP8_XOR1>(t));
    t = fmaxf(t, dpp8f<DPP8_XOR2>(t));
    t = fmaxf(t, dpp8f<DPP8_XOR4>(t));
    t = fmaxf(t, row_mirror(t));
    return t;
}
__device__ __forceinline__ float rowsum16(float t) {
    t += dpp8f<DPP8_XOR1>(t);
    t += dpp8f<DPP8_XOR2>(t);
    t += dpp8f<DPP8_XOR4>(t);
    t += row_mirror(t);
    return t;
}

// Stage one BN x D_ tile of K (row-major f16) and V (transposed f16) into LDS.
__device__ __forceinline__ void stage_tile(const float* __restrict__ Kb,
                                           const float* __restrict__ Vb,
                                           int kv0, int tid,
                                           _Float16* __restrict__ dK,
                                           _Float16* __restrict__ dVt)
{
    const int r  = tid >> 2;          // 0..31 kv row
    const int c0 = (tid & 3) * 32;    // 0,32,64,96 d column
    const float* kp = Kb + (size_t)(kv0 + r) * D_ + c0;
    _Float16* dst = dK + r * D_ + c0;
    #pragma unroll
    for (int j = 0; j < 32; j += 4) {
        float4 f = *(const float4*)(kp + j);
        dst[j + 0] = (_Float16)f.x;
        dst[j + 1] = (_Float16)f.y;
        dst[j + 2] = (_Float16)f.z;
        dst[j + 3] = (_Float16)f.w;
    }
    const float* vp = Vb + (size_t)(kv0 + r) * D_ + c0;
    #pragma unroll
    for (int j = 0; j < 32; j += 4) {
        float4 f = *(const float4*)(vp + j);
        dVt[(c0 + j + 0) * BN + r] = (_Float16)f.x;
        dVt[(c0 + j + 1) * BN + r] = (_Float16)f.y;
        dVt[(c0 + j + 2) * BN + r] = (_Float16)f.z;
        dVt[(c0 + j + 3) * BN + r] = (_Float16)f.w;
    }
}

// Flash-attention forward: one block per (batch, 64-row q-tile).
__global__ __launch_bounds__(128) void fa_fwd_kernel(
    const float* __restrict__ Q,
    const float* __restrict__ K,
    const float* __restrict__ V,
    float* __restrict__ Out)
{
    __shared__ _Float16 sK[2][BN * D_];          // double-buffered K tiles
    __shared__ _Float16 sVt[2][D_ * BN];         // double-buffered V^T tiles
    __shared__ _Float16 sP[NWAVES * 16 * BN];    // per-wave P staging

    const int tid  = threadIdx.x;
    const int wave = tid >> 5;
    const int lane = tid & 31;
    const int n    = lane & 15;   // column / row-in-tile index
    const int g    = lane >> 4;   // half-wave group

    const int batch = blockIdx.x / (SQ_ / BM);
    const int qtile = blockIdx.x % (SQ_ / BM);
    const int qrow0 = qtile * BM + wave * 16;

    const float* Qb = Q + (size_t)batch * SQ_  * D_;
    const float* Kb = K + (size_t)batch * SKV_ * D_;
    const float* Vb = V + (size_t)batch * SKV_ * D_;
    float*       Ob = Out + (size_t)batch * SQ_ * D_;

    // ---- Load Q rows for this wave as 4 A-fragments (16x32 f16 each) ----
    // A operand, contiguous k order (B matches): lane L holds row m=L%16; element j holds
    // K = 16*(L/16) + j  (one contiguous 16-float chunk).
    v16h aQ[4];
    {
        const float* qrow = Qb + (size_t)(qrow0 + n) * D_;
        #pragma unroll
        for (int f = 0; f < 4; ++f) {
            const float* qp = qrow + f * 32 + 16 * g;
            v16h a;
            #pragma unroll
            for (int j = 0; j < 8; ++j) {
                a[j]     = (_Float16)qp[j];
                a[j + 8] = (_Float16)qp[j + 8];
            }
            aQ[f] = a;
        }
    }

    v8f Oacc[8];
    #pragma unroll
    for (int i = 0; i < 8; ++i) Oacc[i] = {};

    float mrow[8], lrow[8];
    #pragma unroll
    for (int r = 0; r < 8; ++r) { mrow[r] = -INFINITY; lrow[r] = 0.0f; }

    // 1/sqrt(128) * log2(e): do softmax in base-2 (v_exp_f32).
    const float qk_scale = 0.08838834764831845f * 1.44269504088896340736f;

    _Float16* sPw = &sP[wave * 16 * BN];

    // Prologue: stage tile 0.
    stage_tile(Kb, Vb, 0, tid, sK[0], sVt[0]);

    int buf = 0;
    for (int it = 0; it < NIT; ++it) {
        __syncthreads();   // staged writes visible; previous reads retired

        // Kick off next tile's global loads into the other buffer; the VMEM
        // latency hides under this iteration's WMMA + softmax work.
        if (it + 1 < NIT)
            stage_tile(Kb, Vb, (it + 1) * BN, tid, sK[buf ^ 1], sVt[buf ^ 1]);

        const _Float16* cK  = sK[buf];
        const _Float16* cVt = sVt[buf];

        // ---- S = Q K^T : two 16x16 f32 tiles, accumulate over D in 4 steps ----
        // B operand, contiguous k order (A matches): lane L holds col n=L%16; element j holds
        // K = 16*(L/16) + j  -> contiguous 16-half chunk of a cK row.
        v8f S0 = {}, S1 = {};
        #pragma unroll
        for (int f = 0; f < 4; ++f) {
            v16h b0 = *(const v16h*)&cK[(0 * 16 + n) * D_ + f * 32 + 16 * g];
            S0 = __builtin_amdgcn_wmma_f32_16x16x32_f16(false, aQ[f], false, b0,
                                                        (short)0, S0, false, false);
            v16h b1 = *(const v16h*)&cK[(1 * 16 + n) * D_ + f * 32 + 16 * g];
            S1 = __builtin_amdgcn_wmma_f32_16x16x32_f16(false, aQ[f], false, b1,
                                                        (short)0, S1, false, false);
        }

        // ---- Online softmax (C layout: vgpr r, group g -> row r+8g, col n) ----
        #pragma unroll
        for (int r = 0; r < 8; ++r) {
            float s0 = S0[r] * qk_scale;
            float s1 = S1[r] * qk_scale;
            float t  = rowmax16(fmaxf(s0, s1));
            float mnew  = fmaxf(mrow[r], t);
            float alpha = exp2_fast(mrow[r] - mnew);
            float p0 = exp2_fast(s0 - mnew);
            float p1 = exp2_fast(s1 - mnew);
            float rs = rowsum16(p0 + p1);
            lrow[r] = lrow[r] * alpha + rs;
            mrow[r] = mnew;
            #pragma unroll
            for (int nb = 0; nb < 8; ++nb) Oacc[nb][r] *= alpha;
            // C-layout -> row-major f16 staging for the A-layout reload
            sPw[(r + 8 * g) * BN + n]      = (_Float16)p0;
            sPw[(r + 8 * g) * BN + 16 + n] = (_Float16)p1;
        }

        // Wave-private region; DS is in-order per wave — a dscnt fence suffices.
        asm volatile("s_wait_dscnt 0" ::: "memory");

        // ---- Reload P as one A-fragment (16x32 f16) ----
        const _Float16* pp = sPw + n * BN + 16 * g;
        v8h plo = *(const v8h*)(pp);
        v8h phi = *(const v8h*)(pp + 8);
        v16h pa = __builtin_shufflevector(plo, phi,
                    0, 1, 2, 3, 4, 5, 6, 7, 8, 9, 10, 11, 12, 13, 14, 15);

        // ---- O += P * V : 8 output d-blocks of 16 columns ----
        #pragma unroll
        for (int nb = 0; nb < 8; ++nb) {
            v16h bv = *(const v16h*)&cVt[(nb * 16 + n) * BN + 16 * g];
            Oacc[nb] = __builtin_amdgcn_wmma_f32_16x16x32_f16(false, pa, false, bv,
                                                              (short)0, Oacc[nb],
                                                              false, false);
        }

        buf ^= 1;
    }

    // ---- Normalize and store ----
    #pragma unroll
    for (int r = 0; r < 8; ++r) {
        float inv = 1.0f / lrow[r];
        float* orow = Ob + (size_t)(qrow0 + r + 8 * g) * D_;
        #pragma unroll
        for (int nb = 0; nb < 8; ++nb) {
            orow[nb * 16 + n] = Oacc[nb][r] * inv;
        }
    }
}

extern "C" void kernel_launch(void* const* d_in, const int* in_sizes, int n_in,
                              void* d_out, int out_size, void* d_ws, size_t ws_size,
                              hipStream_t stream) {
    const float* q = (const float*)d_in[0];
    const float* k = (const float*)d_in[1];
    const float* v = (const float*)d_in[2];
    float* out = (float*)d_out;
    (void)in_sizes; (void)n_in; (void)out_size; (void)d_ws; (void)ws_size;

    dim3 grid(B_ * (SQ_ / BM));   // 16 * 32 = 512 blocks
    dim3 block(128);              // 4 waves (wave32)
    hipLaunchKernelGGL(fa_fwd_kernel, grid, block, 0, stream, q, k, v, out);
}
